// SfBGT_51573967290527
// MI455X (gfx1250) — compile-verified, matrix operand order checked by reading, not run on hardware
//
#include <hip/hip_runtime.h>
#include <hip/hip_bf16.h>
#include <cstdint>
#include <cstddef>

// ---------------------------------------------------------------------------
// Types for CDNA5 WMMA (wave32): A/B = 16 bf16 per lane, C/D = 8 f32 per lane.
// ---------------------------------------------------------------------------
typedef __bf16 bf16;
typedef __attribute__((ext_vector_type(16))) __bf16 v16bf;
typedef __attribute__((ext_vector_type(8)))  float  v8f;

union Frag16 { uint4 u[2]; v16bf v; };   // 32 bytes = one 16-bit A/B fragment
union U8h    { uint4 u; bf16 h[8]; };

// Problem constants (match reference)
constexpr int Lc = 4, Bc = 4, Nc = 1024, Dc = 768, Hc = 12, DHc = 64, Fc = 3072;

// ---------------------------------------------------------------------------
// CDNA5 async global->LDS copy (ASYNCcnt-tracked, bypasses VGPRs).
// ldsOff = LDS byte address for this lane (low 32 bits of flat shared ptr),
// gaddr  = 64-bit global address. 16 bytes per lane.
// ---------------------------------------------------------------------------
__device__ __forceinline__ void async_load_b128(uint32_t ldsOff, uint64_t gaddr) {
  asm volatile("global_load_async_to_lds_b128 %0, %1, off"
               :
               : "v"(ldsOff), "v"(gaddr)
               : "memory");
}
__device__ __forceinline__ void wait_asynccnt0() {
  asm volatile("s_wait_asynccnt 0x0" ::: "memory");
}

// ---------------------------------------------------------------------------
// 16-lane-group reductions (rows of a C-fragment live across 16 lanes)
// ---------------------------------------------------------------------------
__device__ __forceinline__ float sumHalf16(float v) {
  v += __shfl_xor(v, 1, 32); v += __shfl_xor(v, 2, 32);
  v += __shfl_xor(v, 4, 32); v += __shfl_xor(v, 8, 32);
  return v;
}
__device__ __forceinline__ float maxHalf16(float v) {
  v = fmaxf(v, __shfl_xor(v, 1, 32)); v = fmaxf(v, __shfl_xor(v, 2, 32));
  v = fmaxf(v, __shfl_xor(v, 4, 32)); v = fmaxf(v, __shfl_xor(v, 8, 32));
  return v;
}

// ---------------------------------------------------------------------------
// fp32 -> bf16 conversion (weights, per launch; deterministic)
// ---------------------------------------------------------------------------
__global__ __launch_bounds__(256) void f32_to_bf16_k(const float* __restrict__ src,
                                                     bf16* __restrict__ dst, int n) {
  int i = blockIdx.x * 256 + threadIdx.x;
  if (i < n) dst[i] = (bf16)src[i];
}

// ---------------------------------------------------------------------------
// LayerNorm over D=768, output bf16 (pre-LN input for the GEMMs).
// One 256-thread block per row; each thread owns 3 elements.
// ---------------------------------------------------------------------------
__global__ __launch_bounds__(256) void ln_to_bf16_k(const float* __restrict__ h,
                                                    const float* __restrict__ g,
                                                    const float* __restrict__ b,
                                                    bf16* __restrict__ out) {
  int row = blockIdx.x;
  int tid = threadIdx.x;
  const float* x = h + (size_t)row * Dc;
  float v0 = x[tid], v1 = x[tid + 256], v2 = x[tid + 512];
  float s = v0 + v1 + v2;
  float s2 = v0 * v0 + v1 * v1 + v2 * v2;
  #pragma unroll
  for (int m = 16; m >= 1; m >>= 1) {
    s  += __shfl_xor(s,  m, 32);
    s2 += __shfl_xor(s2, m, 32);
  }
  __shared__ float red[16];
  int wave = tid >> 5, lane = tid & 31;
  if (lane == 0) { red[wave] = s; red[8 + wave] = s2; }
  __syncthreads();
  float S = 0.f, S2 = 0.f;
  #pragma unroll
  for (int w = 0; w < 8; w++) { S += red[w]; S2 += red[8 + w]; }
  float mean = S * (1.f / Dc);
  float var  = S2 * (1.f / Dc) - mean * mean;
  float inv  = rsqrtf(var + 1e-5f);
  bf16* o = out + (size_t)row * Dc;
  o[tid]       = (bf16)((v0 - mean) * inv * g[tid]       + b[tid]);
  o[tid + 256] = (bf16)((v1 - mean) * inv * g[tid + 256] + b[tid + 256]);
  o[tid + 512] = (bf16)((v2 - mean) * inv * g[tid + 512] + b[tid + 512]);
}

// ---------------------------------------------------------------------------
// bf16 GEMM: C[M,N] = A[M,K] * B[K,N] + bias  (+ optional fp32 residual)
// Block tile 128x128, K-step 32. 8 waves (2x4); each wave: 64x32 = 4x2 frags.
// Double-buffered LDS: A tile staged with global_load_async_to_lds_b128
// (ASYNCcnt) while WMMAs consume the other buffer; B tile transposed via
// VGPRs (stride 40 halves keeps every fragment load two 16B ds_load_b128).
// EPI==0: write bf16 output. EPI==1: write fp32 resid + acc + bias.
// ---------------------------------------------------------------------------
template <int EPI>
__global__ __launch_bounds__(256) void gemm_bf16_k(
    const bf16* __restrict__ A, const bf16* __restrict__ Bm,
    const float* __restrict__ bias, const float* __restrict__ resid,
    float* __restrict__ outF, bf16* __restrict__ outB,
    int M, int K, int Nn) {
  constexpr int SA = 40, SB = 40;
  __shared__ __align__(16) bf16 sA[2][128 * SA];
  __shared__ __align__(16) bf16 sB[2][128 * SB];

  int tid  = threadIdx.x;
  int wave = tid >> 5, lane = tid & 31;
  int wm = wave >> 2, wn = wave & 3;        // 2 x 4 wave grid
  int lcol = lane & 15, khalf = lane >> 4;  // WMMA lane decomposition
  int m0 = blockIdx.y * 128, n0 = blockIdx.x * 128;

  v8f vz = {0.f, 0.f, 0.f, 0.f, 0.f, 0.f, 0.f, 0.f};
  v8f acc[4][2];
  #pragma unroll
  for (int mf = 0; mf < 4; mf++)
    #pragma unroll
    for (int nf = 0; nf < 2; nf++) acc[mf][nf] = vz;

  int aRow = tid >> 2, aSeg = tid & 3;    // A: 128 rows x 4 x uint4
  int bRow = tid >> 4, bSeg = tid & 15;   // B: 32 rows x 16 x uint4

  auto stage = [&](int buf, int k0) {
    #pragma unroll
    for (int rep = 0; rep < 2; rep++) {
      int r = aRow + rep * 64;
      uint32_t lo = (uint32_t)(uintptr_t)&sA[buf][r * SA + aSeg * 8];
      async_load_b128(lo, (uint64_t)(uintptr_t)(A + (size_t)(m0 + r) * K + k0 + aSeg * 8));
      int kr = bRow + rep * 16;
      U8h db;
      db.u = *(const uint4*)(Bm + (size_t)(k0 + kr) * Nn + n0 + bSeg * 8);
      #pragma unroll
      for (int e = 0; e < 8; e++) sB[buf][(bSeg * 8 + e) * SB + kr] = db.h[e];
    }
    if (k0 + 32 < K)  // keep the B stream warm in L2 (global_prefetch_b8)
      __builtin_prefetch(Bm + (size_t)(k0 + 32 + bRow) * Nn + n0 + bSeg * 8, 0, 0);
  };

  // prologue: tile 0 into buffer 0
  stage(0, 0);
  wait_asynccnt0();
  __syncthreads();

  int nk = K / 32;
  for (int it = 0; it < nk; it++) {
    int cur = it & 1;
    if (it + 1 < nk) stage(cur ^ 1, (it + 1) * 32);  // overlap with compute

    Frag16 bfr[2];
    #pragma unroll
    for (int nf = 0; nf < 2; nf++) {
      int n = wn * 32 + nf * 16 + lcol;
      bfr[nf].u[0] = *(const uint4*)(&sB[cur][n * SB + khalf * 8]);
      bfr[nf].u[1] = *(const uint4*)(&sB[cur][n * SB + 16 + khalf * 8]);
    }
    #pragma unroll
    for (int mf = 0; mf < 4; mf++) {
      int m = wm * 64 + mf * 16 + lcol;
      Frag16 afr;
      afr.u[0] = *(const uint4*)(&sA[cur][m * SA + khalf * 8]);
      afr.u[1] = *(const uint4*)(&sA[cur][m * SA + 16 + khalf * 8]);
      #pragma unroll
      for (int nf = 0; nf < 2; nf++)
        acc[mf][nf] = __builtin_amdgcn_wmma_f32_16x16x32_bf16(
            false, afr.v, false, bfr[nf].v, (short)0, acc[mf][nf], false, false);
    }

    if (it + 1 < nk) wait_asynccnt0();  // own async writes to buf^1 done
    __syncthreads();                    // everyone's staging visible / reads done
  }

  // ---- epilogue: C layout row = r + 8*(lane>>4), col = lane&15 ----
  #pragma unroll
  for (int nf = 0; nf < 2; nf++) {
    int nc = n0 + wn * 32 + nf * 16 + lcol;
    float bvv = bias[nc];
    #pragma unroll
    for (int mf = 0; mf < 4; mf++) {
      #pragma unroll
      for (int r = 0; r < 8; r++) {
        int mr = m0 + wm * 64 + mf * 16 + r + 8 * khalf;
        float val = acc[mf][nf][r] + bvv;
        if (EPI == 0) {
          outB[(size_t)mr * Nn + nc] = (bf16)val;
        } else {
          outF[(size_t)mr * Nn + nc] = resid[(size_t)mr * Nn + nc] + val;
        }
      }
    }
  }
}

// ---------------------------------------------------------------------------
// Flash attention with relation bias: one block = (b, h, 64 q-rows), 4 waves.
// Per wave: 16 q-rows. j-tiles of 32 keys; K tile staged async into LDS
// [j][dh], V tile transposed [dh][j] via VGPRs; P converted C-layout ->
// A-layout through a per-wave LDS tile. Online softmax via 16-lane shfls.
// logits = (Q.K^T)/sqrt(64) + relation_bias[b,i,j] * head_scale[h]
// ---------------------------------------------------------------------------
__global__ __launch_bounds__(128) void flash_attn_k(
    const bf16* __restrict__ qb, const bf16* __restrict__ kb,
    const bf16* __restrict__ vb, const float* __restrict__ rb,
    const float* __restrict__ hsc, bf16* __restrict__ av) {
  constexpr int SK = 72, SV = 40, SP = 40;
  __shared__ __align__(16) bf16 sK[32 * SK];    // [jcol][dh]
  __shared__ __align__(16) bf16 sVt[64 * SV];   // [dh][j]
  __shared__ __align__(16) bf16 sP[4][16 * SP]; // per-wave P tile [qrow][j]

  int tid = threadIdx.x, wave = tid >> 5, lane = tid & 31;
  int lcol = lane & 15, khalf = lane >> 4;
  int hIdx = blockIdx.y, bIdx = blockIdx.z;
  int q0 = blockIdx.x * 64 + wave * 16;
  float hs = hsc[hIdx];
  const float scale = 0.125f;  // 1/sqrt(64)

  // Q fragments (A layout), loaded once: two K-steps of 32 over DH=64
  Frag16 qf[2];
  {
    const bf16* qp = qb + ((size_t)(bIdx * Nc + q0 + lcol) * Dc + hIdx * DHc);
    #pragma unroll
    for (int ks = 0; ks < 2; ks++) {
      qf[ks].u[0] = *(const uint4*)(qp + ks * 32 + khalf * 8);
      qf[ks].u[1] = *(const uint4*)(qp + ks * 32 + 16 + khalf * 8);
    }
  }

  v8f vz = {0.f, 0.f, 0.f, 0.f, 0.f, 0.f, 0.f, 0.f};
  v8f O[4];
  #pragma unroll
  for (int f = 0; f < 4; f++) O[f] = vz;
  float mOld[8], lOld[8];
  #pragma unroll
  for (int r = 0; r < 8; r++) { mOld[r] = -3.0e38f; lOld[r] = 0.f; }

  for (int jt = 0; jt < Nc / 32; jt++) {
    int j0 = jt * 32;
    // ---- stage K async [j][dh]; V transposed [dh][j] via VGPRs ----
    #pragma unroll
    for (int rep = 0; rep < 2; rep++) {
      int id = tid + rep * 128;
      int jr = id >> 2, seg = id & 3;
      size_t gofs = (size_t)(bIdx * Nc + j0 + jr) * Dc + hIdx * DHc + seg * 8;
      uint32_t lo = (uint32_t)(uintptr_t)&sK[jr * SK + seg * 8];
      async_load_b128(lo, (uint64_t)(uintptr_t)(kb + gofs));
      U8h dv;
      dv.u = *(const uint4*)(vb + gofs);
      #pragma unroll
      for (int e = 0; e < 8; e++) sVt[(seg * 8 + e) * SV + jr] = dv.h[e];
    }
    wait_asynccnt0();
    __syncthreads();

    // ---- S = Q * K^T  (two 16x16 column fragments) ----
    v8f sacc[2];
    #pragma unroll
    for (int nf = 0; nf < 2; nf++) {
      int jc = nf * 16 + lcol;
      Frag16 kf0, kf1;
      kf0.u[0] = *(const uint4*)(&sK[jc * SK + khalf * 8]);
      kf0.u[1] = *(const uint4*)(&sK[jc * SK + 16 + khalf * 8]);
      kf1.u[0] = *(const uint4*)(&sK[jc * SK + 32 + khalf * 8]);
      kf1.u[1] = *(const uint4*)(&sK[jc * SK + 48 + khalf * 8]);
      v8f z = vz;
      z = __builtin_amdgcn_wmma_f32_16x16x32_bf16(false, qf[0].v, false, kf0.v,
                                                  (short)0, z, false, false);
      z = __builtin_amdgcn_wmma_f32_16x16x32_bf16(false, qf[1].v, false, kf1.v,
                                                  (short)0, z, false, false);
      sacc[nf] = z;
    }

    // ---- online softmax update ----
    #pragma unroll
    for (int r = 0; r < 8; r++) {
      int i = q0 + r + 8 * khalf;
      size_t rbBase = ((size_t)bIdx * Nc + i) * Nc + j0;
      float l0 = sacc[0][r] * scale + rb[rbBase + lcol] * hs;
      float l1 = sacc[1][r] * scale + rb[rbBase + 16 + lcol] * hs;
      float rm = maxHalf16(fmaxf(l0, l1));
      float mN = fmaxf(mOld[r], rm);
      float p0 = __expf(l0 - mN);
      float p1 = __expf(l1 - mN);
      float rs = sumHalf16(p0 + p1);
      float alpha = __expf(mOld[r] - mN);
      lOld[r] = lOld[r] * alpha + rs;
      mOld[r] = mN;
      #pragma unroll
      for (int f = 0; f < 4; f++) O[f][r] *= alpha;
      bf16* pw = &sP[wave][0];
      pw[(r + 8 * khalf) * SP + lcol]      = (bf16)p0;
      pw[(r + 8 * khalf) * SP + 16 + lcol] = (bf16)p1;
    }

    // ---- O += P * V  (A = P from per-wave LDS, B = V transposed tile) ----
    Frag16 af;
    af.u[0] = *(const uint4*)(&sP[wave][lcol * SP + khalf * 8]);
    af.u[1] = *(const uint4*)(&sP[wave][lcol * SP + 16 + khalf * 8]);
    #pragma unroll
    for (int f = 0; f < 4; f++) {
      int dh = f * 16 + lcol;
      Frag16 vf;
      vf.u[0] = *(const uint4*)(&sVt[dh * SV + khalf * 8]);
      vf.u[1] = *(const uint4*)(&sVt[dh * SV + 16 + khalf * 8]);
      O[f] = __builtin_amdgcn_wmma_f32_16x16x32_bf16(false, af.v, false, vf.v,
                                                     (short)0, O[f], false, false);
    }
    __syncthreads();  // protect K/V tiles before restage
  }

  // ---- finalize: O /= l, write bf16 [B,N,D] ----
  #pragma unroll
  for (int r = 0; r < 8; r++) {
    float inv = 1.f / lOld[r];
    int i = q0 + r + 8 * khalf;
    #pragma unroll
    for (int f = 0; f < 4; f++)
      av[(size_t)(bIdx * Nc + i) * Dc + hIdx * DHc + f * 16 + lcol] =
          (bf16)(O[f][r] * inv);
  }
}

// ---------------------------------------------------------------------------
// GLU: out = val * sigmoid(gate); proj is bf16 [4096, 2F], out bf16 [4096, F]
// ---------------------------------------------------------------------------
__global__ __launch_bounds__(256) void glu_k(const bf16* __restrict__ proj,
                                             bf16* __restrict__ out) {
  int i = blockIdx.x * 256 + threadIdx.x;  // exact multiple: 4096*3072
  int m = i / Fc;
  int n = i - m * Fc;
  float val = (float)proj[(size_t)m * (2 * Fc) + n];
  float gt  = (float)proj[(size_t)m * (2 * Fc) + Fc + n];
  out[i] = (bf16)(val / (1.f + __expf(-gt)));
}

// ---------------------------------------------------------------------------
// Host orchestration
// ---------------------------------------------------------------------------
extern "C" void kernel_launch(void* const* d_in, const int* in_sizes, int n_in,
                              void* d_out, int out_size, void* d_ws, size_t ws_size,
                              hipStream_t stream) {
  const float* hidden = (const float*)d_in[0];
  const float* rb     = (const float*)d_in[1];
  const float* wq     = (const float*)d_in[2];
  const float* bq     = (const float*)d_in[3];
  const float* wk     = (const float*)d_in[4];
  const float* bk     = (const float*)d_in[5];
  const float* wv     = (const float*)d_in[6];
  const float* bv     = (const float*)d_in[7];
  const float* wo     = (const float*)d_in[8];
  const float* bo     = (const float*)d_in[9];
  const float* hsc    = (const float*)d_in[10];
  const float* ln1g   = (const float*)d_in[11];
  const float* ln1b   = (const float*)d_in[12];
  const float* ln2g   = (const float*)d_in[13];
  const float* ln2b   = (const float*)d_in[14];
  const float* w_in   = (const float*)d_in[15];
  const float* b_in   = (const float*)d_in[16];
  const float* w_out  = (const float*)d_in[17];
  const float* b_out  = (const float*)d_in[18];

  float* h = (float*)d_out;  // running hidden state, fp32, updated in place
  hipMemcpyAsync(h, hidden, (size_t)Bc * Nc * Dc * sizeof(float),
                 hipMemcpyDeviceToDevice, stream);

  // workspace carve-up
  char* p = (char*)d_ws;
  auto carve = [&](size_t bytes) -> void* {
    void* r = (void*)p;
    p += (bytes + 255) & ~(size_t)255;
    return r;
  };
  const size_t actB = (size_t)Bc * Nc * Dc * sizeof(bf16);  // 6.29 MB
  bf16* xln   = (bf16*)carve(actB);
  bf16* qbuf  = (bf16*)carve(actB);
  bf16* kbuf  = (bf16*)carve(actB);
  bf16* vbuf  = (bf16*)carve(actB);
  bf16* avb   = (bf16*)carve(actB);
  bf16* proj  = (bf16*)carve((size_t)Bc * Nc * 2 * Fc * sizeof(bf16));  // 50 MB
  bf16* glub  = (bf16*)carve((size_t)Bc * Nc * Fc * sizeof(bf16));      // 25 MB
  bf16* wqb   = (bf16*)carve((size_t)Dc * Dc * sizeof(bf16));
  bf16* wkb   = (bf16*)carve((size_t)Dc * Dc * sizeof(bf16));
  bf16* wvb   = (bf16*)carve((size_t)Dc * Dc * sizeof(bf16));
  bf16* wob   = (bf16*)carve((size_t)Dc * Dc * sizeof(bf16));
  bf16* winb  = (bf16*)carve((size_t)Dc * 2 * Fc * sizeof(bf16));
  bf16* woutb = (bf16*)carve((size_t)Fc * Dc * sizeof(bf16));

  const int M = Bc * Nc;                 // 4096
  dim3 gemmGridD(Dc / 128, M / 128);     // N=768
  dim3 gemmGridF(2 * Fc / 128, M / 128); // N=6144
  auto conv = [&](const float* s, bf16* d, int n) {
    f32_to_bf16_k<<<(n + 255) / 256, 256, 0, stream>>>(s, d, n);
  };

  for (int l = 0; l < Lc; l++) {
    // per-layer weight conversion fp32 -> bf16
    conv(wq    + (size_t)l * Dc * Dc,     wqb,   Dc * Dc);
    conv(wk    + (size_t)l * Dc * Dc,     wkb,   Dc * Dc);
    conv(wv    + (size_t)l * Dc * Dc,     wvb,   Dc * Dc);
    conv(wo    + (size_t)l * Dc * Dc,     wob,   Dc * Dc);
    conv(w_in  + (size_t)l * Dc * 2 * Fc, winb,  Dc * 2 * Fc);
    conv(w_out + (size_t)l * Fc * Dc,     woutb, Fc * Dc);

    // --- relation-biased MHA (pre-LN) ---
    ln_to_bf16_k<<<M, 256, 0, stream>>>(h, ln1g + l * Dc, ln1b + l * Dc, xln);
    gemm_bf16_k<0><<<gemmGridD, 256, 0, stream>>>(xln, wqb, bq + l * Dc,
                                                  nullptr, nullptr, qbuf, M, Dc, Dc);
    gemm_bf16_k<0><<<gemmGridD, 256, 0, stream>>>(xln, wkb, bk + l * Dc,
                                                  nullptr, nullptr, kbuf, M, Dc, Dc);
    gemm_bf16_k<0><<<gemmGridD, 256, 0, stream>>>(xln, wvb, bv + l * Dc,
                                                  nullptr, nullptr, vbuf, M, Dc, Dc);
    flash_attn_k<<<dim3(Nc / 64, Hc, Bc), 128, 0, stream>>>(qbuf, kbuf, vbuf, rb,
                                                            hsc + l * Hc, avb);
    gemm_bf16_k<1><<<gemmGridD, 256, 0, stream>>>(avb, wob, bo + l * Dc,
                                                  h, h, nullptr, M, Dc, Dc);

    // --- GLU feed-forward (pre-LN) ---
    ln_to_bf16_k<<<M, 256, 0, stream>>>(h, ln2g + l * Dc, ln2b + l * Dc, xln);
    gemm_bf16_k<0><<<gemmGridF, 256, 0, stream>>>(xln, winb, b_in + (size_t)l * 2 * Fc,
                                                  nullptr, nullptr, proj, M, Dc, 2 * Fc);
    glu_k<<<(M * Fc) / 256, 256, 0, stream>>>(proj, glub);
    gemm_bf16_k<1><<<gemmGridD, 256, 0, stream>>>(glub, woutb, b_out + l * Dc,
                                                  h, h, nullptr, M, Fc, Dc);
  }
  (void)in_sizes; (void)n_in; (void)out_size; (void)ws_size;
}
